// FitTorch_17746804867402
// MI455X (gfx1250) — matrix operand order repaired and checked by reading, not compiled, not run on hardware
//
#include <hip/hip_runtime.h>
#include <hip/hip_bf16.h>

// ---------------- problem constants (match reference) ----------------
#define NATOMS  50000
#define NCONF   500
#define NROWS   3000000      // 3 * NPAIRS
#define ND      55
#define NDP     64           // padded descriptor stride (16B-aligned rows)
#define HID     64
#define NTILES  (NATOMS / 16)   // 3125, exact

typedef __attribute__((ext_vector_type(16))) _Float16 v16h;
typedef __attribute__((ext_vector_type(8)))  _Float16 v8h;
typedef __attribute__((ext_vector_type(8)))  float    v8f;
typedef __attribute__((ext_vector_type(4)))  unsigned int v4u;
typedef __attribute__((ext_vector_type(4)))  int      v4i;
typedef __attribute__((ext_vector_type(8)))  int      v8i;

// ---------------- WMMA helper (CDNA5 16x16x32 f16 -> f32) ----------------
__device__ __forceinline__ v8f wmma_f16(v16h a, v16h b, v8f c) {
  // 8 args: (neg_a, A, neg_b, B, c_mod, C, reuse_a, reuse_b)
  return __builtin_amdgcn_wmma_f32_16x16x32_f16(
      false, a, false, b, (short)0, c, false, false);
}

// A fragment (16x32 f16) from LDS tile (row stride 72 halves), two ds_read_b128.
// Doc layout: lanes 0-15 -> M=lane, halves 0-7 = K0..7, halves 8-15 = K16..23;
//             lanes 16-31 -> M=lane-16, halves 0-7 = K8..15, halves 8-15 = K24..31.
__device__ __forceinline__ v16h make_afrag(const _Float16* sA, int lane, int kt) {
  int m  = lane & 15;
  int hi = (lane >> 4) << 3;                         // 0 or 8
  const _Float16* row = sA + m * 72 + kt * 32 + hi;  // 16B aligned
  v8h lo = *(const v8h*)row;
  v8h up = *(const v8h*)(row + 16);
  return __builtin_shufflevector(lo, up, 0, 1, 2, 3, 4, 5, 6, 7,
                                 8, 9, 10, 11, 12, 13, 14, 15);
}

// B fragment from fragment-major LDS buffer: lane's 16 halves are contiguous.
__device__ __forceinline__ v16h load_bfrag(const _Float16* sFrag,
                                           int s, int kt, int nt, int lane) {
  const _Float16* p = sFrag + ((((s * 2 + kt) * 4 + nt) * 32) + lane) * 16;
  v8h lo = *(const v8h*)p;
  v8h up = *(const v8h*)(p + 8);
  return __builtin_shufflevector(lo, up, 0, 1, 2, 3, 4, 5, 6, 7,
                                 8, 9, 10, 11, 12, 13, 14, 15);
}

// Branchless weight fetch for the 4 B-operand views (no exec-mask dances):
//   s=0: W1[k][n]  (k<ND)    s=1: W2[k][n]
//   s=2: W2[n][k]            s=3: W1[n][k]  (n<ND)
__device__ __forceinline__ float fetchB(const float* __restrict__ W1,
                                        const float* __restrict__ W2,
                                        int s, int k, int n) {
  bool useW1 = (s == 0) || (s == 3);
  bool tr    = (s >= 2);
  int  r   = tr ? n : k;
  int  c   = tr ? k : n;
  bool in  = (s == 0) ? (k < ND) : ((s == 3) ? (n < ND) : true);
  const float* W = useW1 ? W1 : W2;
  int idx = in ? (r * HID + c) : 0;      // clamp address, load unconditionally
  float v = W[idx];
  return in ? v : 0.f;
}

// 16x64 GEMM tile: 8 WMMAs into acc[4] from staged fragment buffer.
__device__ __forceinline__ void gemm16(const _Float16* sA, const _Float16* sFrag,
                                       int s, int lane, v8f acc[4]) {
  v16h a0 = make_afrag(sA, lane, 0);
  v16h a1 = make_afrag(sA, lane, 1);
#pragma unroll
  for (int nt = 0; nt < 4; ++nt) {
    acc[nt] = wmma_f16(a0, load_bfrag(sFrag, s, 0, nt, lane), acc[nt]);
    acc[nt] = wmma_f16(a1, load_bfrag(sFrag, s, 1, nt, lane), acc[nt]);
  }
}

__device__ __forceinline__ void zero_acc(v8f acc[4]) {
#pragma unroll
  for (int nt = 0; nt < 4; ++nt)
#pragma unroll
    for (int r = 0; r < 8; ++r) acc[nt][r] = 0.f;
}

// ---------------- kernel 0: zero the output buffer ----------------
__global__ void zero_out_kernel(float* __restrict__ p, int n) {
  int i = blockIdx.x * 256 + threadIdx.x;
  if (i < n) p[i] = 0.f;
}

// ---------------- kernel 1: MLP fwd + bwd, one 16-atom tile per wave ----------------
__global__ __launch_bounds__(128) void mlp_fwd_bwd_kernel(
    const float* __restrict__ x, const int* __restrict__ indices,
    const float* __restrict__ W1, const float* __restrict__ B1,
    const float* __restrict__ W2, const float* __restrict__ B2,
    const float* __restrict__ W3, const float* __restrict__ B3,
    float* __restrict__ outE, float* __restrict__ dEdD, int dstride) {
  __shared__ _Float16 sFrag[4 * 8 * 32 * 16];   // 4 views x 8 frags x 32 lanes x 16 halves
  __shared__ _Float16 sAraw[4][16 * 72];

  const int tid  = threadIdx.x;
  const int wave = tid >> 5;
  const int lane = tid & 31;
  const int tile = blockIdx.x * 4 + wave;
  const bool active = (tile < NTILES);          // wave-uniform

  // ---- stage all 4 weight views as f16 fragments (once per block) ----
  for (int e = tid; e < 4 * 8 * 32 * 8; e += 128) {   // 8192 half-pairs
    int hp    = (e & 7) << 1;                  // 0,2,...,14
    int lane2 = (e >> 3) & 31;
    int f     = (e >> 8) & 7;                  // kt*4 + nt
    int s     = (e >> 11);
    int kt = f >> 2, nt = f & 3;
    int kg = (lane2 >> 4) << 3;
    int nf = nt * 16 + (lane2 & 15);
    int k0 = kt * 32 + kg + (hp & 7) + ((hp >> 3) << 4);
    float v0 = fetchB(W1, W2, s, k0,     nf);
    float v1 = fetchB(W1, W2, s, k0 + 1, nf);
    union { _Float16 h[2]; unsigned u; } pk;
    pk.h[0] = (_Float16)v0;
    pk.h[1] = (_Float16)v1;
    *(unsigned*)&sFrag[((s * 8 + f) * 32 + lane2) * 16 + hp] = pk.u;
  }
  __syncthreads();
  if (!active) return;

  const int base = tile * 16;
  _Float16* sA = sAraw[wave];
  const int nl  = lane & 15;
  const int hi8 = (lane >> 4) << 3;            // C-layout M offset (0 or 8)

  // ---- stage x tile as f16, zero-padded to K=64 (branchless) ----
  for (int idx = lane; idx < 16 * 64; idx += 32) {
    int m = idx >> 6, d = idx & 63;
    bool in = (d < ND);
    float v = x[(base + m) * ND + (in ? d : 0)];
    sA[m * 72 + d] = (_Float16)(in ? v : 0.f);
  }
  __builtin_amdgcn_wave_barrier();

  v8f acc[4];
  float t1reg[4][8];                           // (1 - h1^2), kept in C-layout

  // ---- layer 1: h1 = tanh(x @ W1 + b1) ----
  zero_acc(acc);
  gemm16(sA, sFrag, 0, lane, acc);
  __builtin_amdgcn_wave_barrier();
#pragma unroll
  for (int nt = 0; nt < 4; ++nt) {
    int n = nt * 16 + nl;
    float bb = B1[n];
#pragma unroll
    for (int r = 0; r < 8; ++r) {
      float h = tanhf(acc[nt][r] + bb);
      t1reg[nt][r] = 1.f - h * h;
      sA[(r + hi8) * 72 + n] = (_Float16)h;    // h1 -> A of layer 2
    }
  }
  __builtin_amdgcn_wave_barrier();

  // ---- layer 2: h2 = tanh(h1 @ W2 + b2); energy + g2 in the epilogue ----
  zero_acc(acc);
  gemm16(sA, sFrag, 1, lane, acc);
  __builtin_amdgcn_wave_barrier();
  float ep[8];
#pragma unroll
  for (int r = 0; r < 8; ++r) ep[r] = 0.f;
#pragma unroll
  for (int nt = 0; nt < 4; ++nt) {
    int n = nt * 16 + nl;
    float bb = B2[n];
    float w3 = W3[n];
#pragma unroll
    for (int r = 0; r < 8; ++r) {
      float h = tanhf(acc[nt][r] + bb);
      ep[r] += h * w3;                         // per-atom energy partial
      float g2 = (1.f - h * h) * w3;           // cotangent thru layer3 + tanh'
      sA[(r + hi8) * 72 + n] = (_Float16)g2;   // g2 -> A of bwd GEMM
    }
  }
  // energy: reduce over the 16 lanes sharing each M-half, scatter per config
#pragma unroll
  for (int r = 0; r < 8; ++r) {
    float v = ep[r];
    v += __shfl_xor(v, 1, 32);
    v += __shfl_xor(v, 2, 32);
    v += __shfl_xor(v, 4, 32);
    v += __shfl_xor(v, 8, 32);
    ep[r] = v;
  }
  if (nl == 0) {
    float bb3 = B3[0];
#pragma unroll
    for (int r = 0; r < 8; ++r) {
      int atom = base + r + hi8;
      atomicAdd(&outE[indices[atom]], ep[r] + bb3);
    }
  }
  __builtin_amdgcn_wave_barrier();

  // ---- bwd layer 2: g1 = (g2 @ W2^T) * (1 - h1^2) ----
  zero_acc(acc);
  gemm16(sA, sFrag, 2, lane, acc);
  __builtin_amdgcn_wave_barrier();
#pragma unroll
  for (int nt = 0; nt < 4; ++nt) {
    int n = nt * 16 + nl;
#pragma unroll
    for (int r = 0; r < 8; ++r)
      sA[(r + hi8) * 72 + n] = (_Float16)(acc[nt][r] * t1reg[nt][r]);
  }
  __builtin_amdgcn_wave_barrier();

  // ---- bwd layer 1: dEdD = g1 @ W1^T ----
  zero_acc(acc);
  gemm16(sA, sFrag, 3, lane, acc);
#pragma unroll
  for (int nt = 0; nt < 4; ++nt) {
    int d = nt * 16 + nl;
    if (d < ND) {
#pragma unroll
      for (int r = 0; r < 8; ++r)
        dEdD[(base + r + hi8) * dstride + d] = acc[nt][r];
    }
  }
}

// ---------------- kernel 2: forces ----------------
// 256 rows/block. The contiguous xd slab (up to 56320B) is staged into LDS by a
// single Tensor Data Mover descriptor (TENSOR_LOAD_TO_LDS) issued by wave 0:
// 1-D tile of `tile_dim0` 4B elements, tensor_dim0 = valid elements so the TDM's
// OOB rule zero-fills the partial last block. Then one row-dot per thread
// against the L2-resident gathered dEdD, one f32 atomic per row.
__global__ __launch_bounds__(256) void force_kernel(
    const float* __restrict__ xd, const int* __restrict__ xd_indx,
    const int* __restrict__ unique_j, const float* __restrict__ dEdD,
    float* __restrict__ outF, int dstride) {
  __shared__ float sxd[256 * ND];              // 56320B, row stride 55 (gcd(55,64)=1)
  const int tid = threadIdx.x;
  const long rb = (long)blockIdx.x * 256;
  const long rows = (NROWS - rb < 256) ? (NROWS - rb) : 256;
  const int  valid = (int)(rows * ND);         // valid 4B elements in this slab
  const float* src = xd + rb * ND;

  if ((tid >> 5) == 0) {                       // one wave issues the DMA
    unsigned long long ga = (unsigned long long)src;
    unsigned lds0 = (unsigned)(unsigned long long)(void*)sxd; // low 32b = LDS offset

    // D# group 0: count=1 | lds_addr | global_addr[56:0] | type=2 ("image")
    v4u g0;
    g0[0] = 1u;                                        // count=1, user descriptor
    g0[1] = lds0;                                      // lds_addr (bytes)
    g0[2] = (unsigned)(ga & 0xFFFFFFFFu);              // global_addr[31:0]
    g0[3] = (unsigned)((ga >> 32) & 0x01FFFFFFu)       // global_addr[56:32]
          | (2u << 30);                                // type=2

    // D# group 1: data_size=4B, 1-D tile
    v8i g1;
    g1[0] = (int)(2u << 16);                           // data_size=2 (4 bytes)
    g1[1] = (int)(((unsigned)valid & 0xFFFFu) << 16);  // tensor_dim0[15:0] @bits63:48
    g1[2] = (int)(((unsigned)valid >> 16)              // tensor_dim0[31:16]
          | (1u << 16));                               // tensor_dim1 = 1
    g1[3] = (int)((unsigned)(256 * ND) << 16);         // tile_dim0 = 14080 @bits127:112
    g1[4] = 0;                                         // tile_dim1=0, tile_dim2=0
    g1[5] = 256 * ND;                                  // tensor_dim0_stride[31:0]
    g1[6] = 0;                                         // stride hi / dim1_stride lo
    g1[7] = 0;

    v4i gz4 = {0, 0, 0, 0};                            // groups 2/3 unused (<=2D)
    v8i gz8 = {0, 0, 0, 0, 0, 0, 0, 0};                // extra group (6-arg builtin)
    __builtin_amdgcn_tensor_load_to_lds(g0, g1, gz4, gz4, gz8, 0);
    __builtin_amdgcn_s_wait_tensorcnt(0);              // TDM done before barrier
  }
  __syncthreads();

  long r = rb + tid;
  if (r < NROWS) {
    int i   = xd_indx[r * 3 + 0];
    int dim = xd_indx[r * 3 + 2];
    int j   = unique_j[r];
    const float* xr   = sxd + tid * ND;
    const float* drow = dEdD + (long)i * dstride;
    float s = 0.f;
    if (dstride == NDP) {
      const float4* d4 = (const float4*)drow;  // 16B-aligned rows
#pragma unroll
      for (int u = 0; u < 13; ++u) {           // cols 0..51
        float4 q = d4[u];
        s += q.x * xr[4 * u + 0] + q.y * xr[4 * u + 1] +
             q.z * xr[4 * u + 2] + q.w * xr[4 * u + 3];
      }
      s += drow[52] * xr[52] + drow[53] * xr[53] + drow[54] * xr[54];
    } else {
#pragma unroll 5
      for (int d = 0; d < ND; ++d) s += xr[d] * drow[d];
    }
    atomicAdd(&outF[(long)j * 3 + dim], s);
  }
}

// ---------------- host launcher ----------------
extern "C" void kernel_launch(void* const* d_in, const int* in_sizes, int n_in,
                              void* d_out, int out_size, void* d_ws, size_t ws_size,
                              hipStream_t stream) {
  const float* x        = (const float*)d_in[0];
  const float* xd       = (const float*)d_in[1];
  const int*   indices  = (const int*)d_in[2];
  // d_in[3] atoms_per_structure, d_in[4] types: unused
  const int*   xd_indx  = (const int*)d_in[5];
  const int*   unique_j = (const int*)d_in[6];
  const float* W1 = (const float*)d_in[7];
  const float* B1 = (const float*)d_in[8];
  const float* W2 = (const float*)d_in[9];
  const float* B2 = (const float*)d_in[10];
  const float* W3 = (const float*)d_in[11];
  const float* B3 = (const float*)d_in[12];

  float* outE = (float*)d_out;                 // [NCONF]
  float* outF = outE + NCONF;                  // [3*NATOMS]
  float* dEdD = (float*)d_ws;

  // deterministic scratch layout choice from ws_size
  int dstride = (ws_size >= (size_t)NATOMS * NDP * sizeof(float)) ? NDP : ND;

  int nout = NCONF + 3 * NATOMS;               // 150500
  zero_out_kernel<<<(nout + 255) / 256, 256, 0, stream>>>((float*)d_out, nout);

  mlp_fwd_bwd_kernel<<<(NTILES + 3) / 4, 128, 0, stream>>>(
      x, indices, W1, B1, W2, B2, W3, B3, outE, dEdD, dstride);

  force_kernel<<<(NROWS + 255) / 256, 256, 0, stream>>>(
      xd, xd_indx, unique_j, dEdD, outF, dstride);
}
